// AttentionBlock_16853451669900
// MI455X (gfx1250) — matrix operand order repaired and checked by reading, not compiled, not run on hardware
//
#include <hip/hip_runtime.h>

// ---------------------------------------------------------------------------
// Transformer block (pre-LN, causal attention, SiLU MLP) for MI455X / gfx1250.
// bf16 WMMA (v_wmma_f32_16x16x32_bf16) for all matmuls, f32 accumulate.
// B=2, L=2048, E=768, H=12, D=64, F=3072.
// ---------------------------------------------------------------------------

typedef __bf16 bf16_t;
typedef __attribute__((ext_vector_type(16))) __bf16 v16bf;
typedef __attribute__((ext_vector_type(2)))  __bf16 bf16x2;
typedef __attribute__((ext_vector_type(8)))  float  v8f;

#define BATCH 2
#define SEQ   2048
#define EMB   768
#define NH    12
#define HD    64
#define E3    2304     // 3*EMB
#define FF    3072     // 4*EMB

static __device__ inline v8f vzero8() {
  v8f v;
#pragma unroll
  for (int i = 0; i < 8; ++i) v[i] = 0.0f;
  return v;
}

static __device__ inline v8f wmma_bf16(v16bf a, v16bf b, v8f c) {
  return __builtin_amdgcn_wmma_f32_16x16x32_bf16(
      /*neg_a=*/false, a, /*neg_b=*/false, b,
      /*c_mod=*/(short)0, c, /*reuse_a=*/false, /*reuse_b=*/false);
}

// A-fragment: 16x32 (MxK) row-major tile at p, leading dim ld (elements).
// Contiguous k layout used here: lanes 0-15 row m=lane, VGPR r holds K pair
//   k = 16*half + 2*r   (half = lane>=16)
static __device__ inline v16bf load_a_frag(const bf16_t* p, int ld) {
  int lane = threadIdx.x & 31;
  int half = lane >> 4, m = lane & 15;
  const bf16_t* row = p + (size_t)m * ld;
  v16bf a;
#pragma unroll
  for (int r = 0; r < 8; ++r) {
    int k = (half << 4) + (r << 1);
    bf16x2 t = *reinterpret_cast<const bf16x2*>(row + k);
    a[2 * r] = t.x; a[2 * r + 1] = t.y;
  }
  return a;
}

// B-fragment: 32x16 (KxN). Source is B-transposed (row n holds K), ld = elems.
// Contiguous k layout (same as A): lanes 0-15 hold K=0..15, lanes 16-31 hold K=16..31, 2 K / VGPR.
static __device__ inline v16bf load_b_frag(const bf16_t* pT, int ld) {
  int lane = threadIdx.x & 31;
  int half = lane >> 4, n = lane & 15;
  const bf16_t* row = pT + (size_t)n * ld + (half << 4);
  v16bf b;
#pragma unroll
  for (int r = 0; r < 8; ++r) {
    bf16x2 t = *reinterpret_cast<const bf16x2*>(row + 2 * r);
    b[2 * r] = t.x; b[2 * r + 1] = t.y;
  }
  return b;
}

static __device__ inline float rmax16(float v) {
  v = fmaxf(v, __shfl_xor(v, 1));
  v = fmaxf(v, __shfl_xor(v, 2));
  v = fmaxf(v, __shfl_xor(v, 4));
  v = fmaxf(v, __shfl_xor(v, 8));
  return v;
}
static __device__ inline float rsum16(float v) {
  v += __shfl_xor(v, 1);
  v += __shfl_xor(v, 2);
  v += __shfl_xor(v, 4);
  v += __shfl_xor(v, 8);
  return v;
}

// --------------------------- weight transpose ------------------------------
// w[K][N] (fp32 row-major) -> wt[N][K] (bf16 row-major)
__global__ void wtrans_kernel(const float* __restrict__ w, bf16_t* __restrict__ wt,
                              int K, int N) {
  int idx = blockIdx.x * 256 + threadIdx.x;
  if (idx < N * K) {
    int n = idx / K, k = idx - n * K;
    wt[idx] = (bf16_t)w[(size_t)k * N + n];
  }
}

// ------------------------------ layer norm ---------------------------------
// one wave per row of EMB=768; writes bf16
__global__ __launch_bounds__(256) void ln_kernel(const float* __restrict__ x,
                                                 const float* __restrict__ sc,
                                                 const float* __restrict__ bi,
                                                 bf16_t* __restrict__ out) {
  int w = threadIdx.x >> 5, lane = threadIdx.x & 31;
  int row = blockIdx.x * 8 + w;
  const float* xr = x + (size_t)row * EMB;
  float xs[24], s = 0.f, s2 = 0.f;
#pragma unroll
  for (int i = 0; i < 24; ++i) {
    float v = xr[lane + 32 * i];
    xs[i] = v; s += v; s2 += v * v;
  }
#pragma unroll
  for (int m = 1; m < 32; m <<= 1) { s += __shfl_xor(s, m); s2 += __shfl_xor(s2, m); }
  float mu = s * (1.f / EMB);
  float var = s2 * (1.f / EMB) - mu * mu;
  float rstd = rsqrtf(var + 1e-6f);
  bf16_t* orow = out + (size_t)row * EMB;
#pragma unroll
  for (int i = 0; i < 24; ++i) {
    int e = lane + 32 * i;
    orow[e] = (bf16_t)((xs[i] - mu) * rstd * sc[e] + bi[e]);
  }
}

// ------------------------------ WMMA GEMM ----------------------------------
// C[M,N] = A[M,K] (bf16) * Bt[N,K]^T (bf16) + bias, with epilogue:
//   EPI 0: write bf16 (qkv) + scatter transposed V copy
//   EPI 1: fp32 out = resid + val
//   EPI 2: bf16 out = silu(val)
// block = 256 thr = 8 waves; block tile 32(M) x 256(N); wave tile 16 x 64.
template <int EPI>
__global__ __launch_bounds__(256) void gemm_wmma_kernel(
    const bf16_t* __restrict__ A, const bf16_t* __restrict__ Bt,
    const float* __restrict__ bias, const float* __restrict__ resid,
    float* __restrict__ outf, bf16_t* __restrict__ outb,
    bf16_t* __restrict__ vt, int M, int N, int K) {
  int w = threadIdx.x >> 5, lane = threadIdx.x & 31;
  int half = lane >> 4, nn = lane & 15;
  int m0 = blockIdx.y * 32 + (w >> 2) * 16;
  int n0 = blockIdx.x * 256 + (w & 3) * 64;

  v8f acc[4];
#pragma unroll
  for (int j = 0; j < 4; ++j) acc[j] = vzero8();

  const bf16_t* arow = A + (size_t)m0 * K;
  for (int k0 = 0; k0 < K; k0 += 32) {
    v16bf af = load_a_frag(arow + k0, K);
#pragma unroll
    for (int j = 0; j < 4; ++j) {
      v16bf bf_ = load_b_frag(Bt + (size_t)(n0 + 16 * j) * K + k0, K);
      acc[j] = wmma_bf16(af, bf_, acc[j]);
    }
  }

#pragma unroll
  for (int j = 0; j < 4; ++j) {
    int n = n0 + 16 * j + nn;
    float bv = bias[n];
#pragma unroll
    for (int r = 0; r < 8; ++r) {
      int m = m0 + r + 8 * half;
      float val = acc[j][r] + bv;
      size_t idx = (size_t)m * N + n;
      if (EPI == 0) {
        outb[idx] = (bf16_t)val;
        if (n >= 2 * EMB) {  // scatter transposed V: vt[b][h][d][l]
          int f = n - 2 * EMB;
          int hh = f >> 6, d = f & 63;
          int bb = m >> 11, l = m & (SEQ - 1);
          vt[(((size_t)(bb * NH + hh)) * HD + d) * SEQ + l] = (bf16_t)val;
        }
      } else if (EPI == 1) {
        outf[idx] = resid[idx] + val;
      } else {
        float sg = val / (1.f + __expf(-val));  // silu
        outb[idx] = (bf16_t)sg;
      }
    }
  }
}

// --------------------------- flash attention -------------------------------
// One wave per (b, h, 16-query tile). qkv bf16 [B*L][3E]; vt bf16 [B][H][D][L].
__global__ __launch_bounds__(256) void attn_kernel(const bf16_t* __restrict__ qkv,
                                                   const bf16_t* __restrict__ vt,
                                                   bf16_t* __restrict__ yb) {
  __shared__ __align__(16) bf16_t sP[8][16 * 32];
  int w = threadIdx.x >> 5, lane = threadIdx.x & 31;
  int half = lane >> 4, nn = lane & 15;
  int g = blockIdx.x * 8 + w;
  int b = g / (NH * (SEQ / 16));
  int rem = g - b * NH * (SEQ / 16);
  int h = rem >> 7;           // SEQ/16 = 128
  int qt = rem & 127;
  int m0 = qt << 4;

  const bf16_t* qbase = qkv + ((size_t)(b * SEQ + m0)) * E3 + h * HD;
  v16bf aq0 = load_a_frag(qbase, E3);
  v16bf aq1 = load_a_frag(qbase + 32, E3);

  const bf16_t* kbase = qkv + (size_t)b * SEQ * E3 + EMB + h * HD;
  const bf16_t* vbase = vt + ((size_t)(b * NH + h)) * HD * SEQ;

  v8f acc[4];
#pragma unroll
  for (int j = 0; j < 4; ++j) acc[j] = vzero8();
  float mi[8], li[8];
#pragma unroll
  for (int r = 0; r < 8; ++r) { mi[r] = -1e30f; li[r] = 0.f; }

  for (int k0 = 0; k0 < m0 + 16; k0 += 32) {
    // S = Q * K^T for 32 keys -> two 16x16 tiles, D=64 -> 2 k-steps each
    v8f s0 = vzero8(), s1 = vzero8();
    const bf16_t* kptr = kbase + (size_t)k0 * E3;
    s0 = wmma_bf16(aq0, load_b_frag(kptr, E3), s0);
    s0 = wmma_bf16(aq1, load_b_frag(kptr + 32, E3), s0);
    kptr += (size_t)16 * E3;
    s1 = wmma_bf16(aq0, load_b_frag(kptr, E3), s1);
    s1 = wmma_bf16(aq1, load_b_frag(kptr + 32, E3), s1);

    // online softmax (scale 1/sqrt(64)=0.125, causal mask)
#pragma unroll
    for (int r = 0; r < 8; ++r) {
      int mrow = m0 + r + 8 * half;
      int c0 = k0 + nn, c1 = c0 + 16;
      float v0 = (c0 <= mrow) ? s0[r] * 0.125f : -1e30f;
      float v1 = (c1 <= mrow) ? s1[r] * 0.125f : -1e30f;
      float nm = fmaxf(mi[r], rmax16(fmaxf(v0, v1)));
      float p0 = __expf(v0 - nm), p1 = __expf(v1 - nm);
      float rs = rsum16(p0 + p1);
      float alpha = __expf(mi[r] - nm);
      li[r] = li[r] * alpha + rs;
      mi[r] = nm;
      acc[0][r] *= alpha; acc[1][r] *= alpha;
      acc[2][r] *= alpha; acc[3][r] *= alpha;
      sP[w][(r + 8 * half) * 32 + nn] = (bf16_t)p0;
      sP[w][(r + 8 * half) * 32 + nn + 16] = (bf16_t)p1;
    }
    asm volatile("s_wait_dscnt 0" ::: "memory");  // wave-local LDS ordering

    // O += P * V   (P: 16x32 via LDS re-fragmentation; V^T contiguous in K)
    v16bf ap = load_a_frag(&sP[w][0], 32);
#pragma unroll
    for (int j = 0; j < 4; ++j) {
      v16bf bv = load_b_frag(vbase + (size_t)(16 * j) * SEQ + k0, SEQ);
      acc[j] = wmma_bf16(ap, bv, acc[j]);
    }
  }

#pragma unroll
  for (int j = 0; j < 4; ++j)
#pragma unroll
    for (int r = 0; r < 8; ++r) {
      int m = m0 + r + 8 * half;
      float o = acc[j][r] / li[r];
      yb[((size_t)(b * SEQ + m)) * EMB + h * HD + 16 * j + nn] = (bf16_t)o;
    }
}

// mask_len handling: query rows >= mask_len have mask==0 for ALL keys ->
// softmax over equal values -> uniform 1/L -> y row = mean over L of V.
__global__ void meanv_kernel(const bf16_t* __restrict__ qkv, float* __restrict__ mv) {
  int idx = blockIdx.x * 256 + threadIdx.x;  // B*EMB = 1536
  if (idx < BATCH * EMB) {
    int b = idx / EMB, f = idx - b * EMB;
    const bf16_t* p = qkv + (size_t)b * SEQ * E3 + 2 * EMB + f;
    float s = 0.f;
    for (int l = 0; l < SEQ; ++l) s += (float)p[(size_t)l * E3];
    mv[idx] = s * (1.f / SEQ);
  }
}

__global__ void fixup_kernel(const int* __restrict__ mask_len,
                             const float* __restrict__ mv,
                             bf16_t* __restrict__ yb) {
  int m = blockIdx.x;
  int b = m >> 11, l = m & (SEQ - 1);
  if (l >= mask_len[b]) {
    for (int f = threadIdx.x; f < EMB; f += 256)
      yb[(size_t)m * EMB + f] = (bf16_t)mv[b * EMB + f];
  }
}

// ---------------------------------------------------------------------------
extern "C" void kernel_launch(void* const* d_in, const int* in_sizes, int n_in,
                              void* d_out, int out_size, void* d_ws, size_t ws_size,
                              hipStream_t stream) {
  const float* x        = (const float*)d_in[0];
  const float* ln1_s    = (const float*)d_in[1];
  const float* ln1_b    = (const float*)d_in[2];
  const float* w_qkv    = (const float*)d_in[3];
  const float* b_qkv    = (const float*)d_in[4];
  const float* w_proj   = (const float*)d_in[5];
  const float* b_proj   = (const float*)d_in[6];
  const float* ln2_s    = (const float*)d_in[7];
  const float* ln2_b    = (const float*)d_in[8];
  const float* w_fc     = (const float*)d_in[9];
  const float* b_fc     = (const float*)d_in[10];
  const float* w_out    = (const float*)d_in[11];
  const float* b_out    = (const float*)d_in[12];
  const int*   mask_len = (const int*)d_in[13];
  float* out = (float*)d_out;

  const int BL = BATCH * SEQ;  // 4096
  size_t off = 0;
  auto alloc = [&](size_t bytes) {
    size_t o = off; off += (bytes + 255) & ~(size_t)255; return (char*)d_ws + o;
  };
  bf16_t* zb      = (bf16_t*)alloc((size_t)BL * EMB * 2);
  bf16_t* wqkv_t  = (bf16_t*)alloc((size_t)E3 * EMB * 2);
  bf16_t* wproj_t = (bf16_t*)alloc((size_t)EMB * EMB * 2);
  bf16_t* wfc_t   = (bf16_t*)alloc((size_t)FF * EMB * 2);
  bf16_t* wout_t  = (bf16_t*)alloc((size_t)EMB * FF * 2);
  bf16_t* qkvb    = (bf16_t*)alloc((size_t)BL * E3 * 2);
  bf16_t* vt      = (bf16_t*)alloc((size_t)BATCH * EMB * SEQ * 2);
  bf16_t* yb      = (bf16_t*)alloc((size_t)BL * EMB * 2);
  float*  x1      = (float*)alloc((size_t)BL * EMB * 4);
  bf16_t* hb      = (bf16_t*)alloc((size_t)BL * FF * 2);
  float*  mv      = (float*)alloc((size_t)BATCH * EMB * 4);
  (void)ws_size; (void)n_in; (void)in_sizes; (void)out_size;

  // 1) weights -> bf16 transposed [N][K]
  wtrans_kernel<<<(EMB * E3 + 255) / 256, 256, 0, stream>>>(w_qkv, wqkv_t, EMB, E3);
  wtrans_kernel<<<(EMB * EMB + 255) / 256, 256, 0, stream>>>(w_proj, wproj_t, EMB, EMB);
  wtrans_kernel<<<(EMB * FF + 255) / 256, 256, 0, stream>>>(w_fc, wfc_t, EMB, FF);
  wtrans_kernel<<<(FF * EMB + 255) / 256, 256, 0, stream>>>(w_out, wout_t, FF, EMB);

  // 2) LN1
  ln_kernel<<<BL / 8, 256, 0, stream>>>(x, ln1_s, ln1_b, zb);

  // 3) qkv = z @ w_qkv + b_qkv  (also writes V^T)
  gemm_wmma_kernel<0><<<dim3(E3 / 256, BL / 32), 256, 0, stream>>>(
      zb, wqkv_t, b_qkv, nullptr, nullptr, qkvb, vt, BL, E3, EMB);

  // 4) causal flash attention
  attn_kernel<<<(BATCH * NH * (SEQ / 16)) / 8, 256, 0, stream>>>(qkvb, vt, yb);

  // 5) mask_len rows -> uniform attention = mean(V)
  meanv_kernel<<<(BATCH * EMB + 255) / 256, 256, 0, stream>>>(qkvb, mv);
  fixup_kernel<<<BL, 256, 0, stream>>>(mask_len, mv, yb);

  // 6) x1 = x + y @ w_proj + b_proj
  gemm_wmma_kernel<1><<<dim3(EMB / 256, BL / 32), 256, 0, stream>>>(
      yb, wproj_t, b_proj, x, x1, nullptr, nullptr, BL, EMB, EMB);

  // 7) LN2
  ln_kernel<<<BL / 8, 256, 0, stream>>>(x1, ln2_s, ln2_b, zb);

  // 8) h = silu(z2 @ w_fc + b_fc)
  gemm_wmma_kernel<2><<<dim3(FF / 256, BL / 32), 256, 0, stream>>>(
      zb, wfc_t, b_fc, nullptr, nullptr, hb, nullptr, BL, FF, EMB);

  // 9) out = x1 + h @ w_out + b_out
  gemm_wmma_kernel<1><<<dim3(EMB / 256, BL / 32), 256, 0, stream>>>(
      hb, wout_t, b_out, x1, out, nullptr, nullptr, BL, EMB, FF);
}
